// Model_33260226740708
// MI455X (gfx1250) — matrix operand order repaired and checked by reading, not compiled, not run on hardware
//
#include <hip/hip_runtime.h>
#include <hip/hip_bf16.h>

typedef _Float16 half_t;
typedef __attribute__((ext_vector_type(16))) _Float16 v16h;
typedef __attribute__((ext_vector_type(8)))  _Float16 v8h;
typedef __attribute__((ext_vector_type(8)))  float    v8f;

#define CB   128      // batch
#define CT   600      // timesteps
#define CU   96       // attention window length
#define CK   10       // mixture components
#define CCD  80       // c_vec feature dim
#define CS   512      // hidden size
#define COUT 121      // 1 + 6*M
#define D1P  608      // 3+80+512 = 595 padded to 32
#define D2P  1120     // 3+512+80+512 = 1107 padded to 32
#define NWG  32       // persistent workgroups (256 waves total)

// ---------------------------------------------------------------- device utils

__device__ __forceinline__ float sigm(float x) { return 1.0f / (1.0f + __expf(-x)); }

__device__ __forceinline__ v8f wmma32(v16h a, v16h b, v8f c) {
  return __builtin_amdgcn_wmma_f32_16x16x32_f16(false, a, false, b, (short)0, c,
                                                false, false);
}

// A: row-major [M x K] f16; BT: row-major [N x K] f16 (i.e. B transposed).
// Computes two 16x16 f32 accumulators: columns nJ..nJ+15 and nK..nK+15.
__device__ __forceinline__ void gemm_pair(const half_t* __restrict__ A, int lda,
                                          const half_t* __restrict__ BT, int ldb,
                                          int m0, int nJ, int nK, int kIters,
                                          int lane, v8f& cJ, v8f& cK) {
  const int q = lane & 15, hf = lane >> 4;
  const half_t* ap = A  + (size_t)(m0 + q) * lda + 16 * hf;
  const half_t* bj = BT + (size_t)(nJ + q) * ldb + 16 * hf;
  const half_t* bk = BT + (size_t)(nK + q) * ldb + 16 * hf;
  for (int kk = 0; kk < kIters; ++kk) {
    const int k0 = kk * 32;
    union { v16h v; v8h p[2]; } a;
    a.p[0] = *(const v8h*)(ap + k0);        // K = k0 + 16*half .. +7
    a.p[1] = *(const v8h*)(ap + k0 + 8);    // K = k0 + 16*half + 8 .. +15
    const v16h vbj = *(const v16h*)(bj + k0);
    const v16h vbk = *(const v16h*)(bk + k0);
    cJ = wmma32(a.v, vbj, cJ);
    cK = wmma32(a.v, vbk, cK);
  }
}

__device__ __forceinline__ void gemm_one(const half_t* __restrict__ A, int lda,
                                         const half_t* __restrict__ BT, int ldb,
                                         int m0, int n0, int kIters,
                                         int lane, v8f& c) {
  const int q = lane & 15, hf = lane >> 4;
  const half_t* ap = A  + (size_t)(m0 + q) * lda + 16 * hf;
  const half_t* bp = BT + (size_t)(n0 + q) * ldb + 16 * hf;
  for (int kk = 0; kk < kIters; ++kk) {
    const int k0 = kk * 32;
    union { v16h v; v8h p[2]; } a;
    a.p[0] = *(const v8h*)(ap + k0);
    a.p[1] = *(const v8h*)(ap + k0 + 8);
    const v16h vb = *(const v16h*)(bp + k0);
    c = wmma32(a.v, vb, c);
  }
}

// Gate epilogue: h_new = sig(j)*(1-h) + (1-sig(k))*h, broadcast f16 copies.
__device__ __forceinline__ void gate_epilogue(
    const v8f& cJ, const v8f& cK,
    const float* __restrict__ bjx, const float* __restrict__ bjh,
    const float* __restrict__ bkx, const float* __restrict__ bkh,
    float* __restrict__ hstate, int m0, int n0, int lane,
    half_t* __restrict__ dst0, int ld0, int off0,
    half_t* __restrict__ dst1, int ld1, int off1,
    half_t* __restrict__ dst2, int ld2, int off2) {
  const int n = lane & 15, rb = (lane >> 4) * 8;
  const int s = n0 + n;
  const float bj = bjx[s] + bjh[s];
  const float bk = bkx[s] + bkh[s];
#pragma unroll
  for (int v = 0; v < 8; ++v) {
    const int b = m0 + rb + v;
    const float jg = sigm(cJ[v] + bj);
    const float kg = sigm(cK[v] + bk);
    float* hp = hstate + (size_t)b * CS + s;
    const float ho = *hp;
    const float hn = jg * (1.0f - ho) + (1.0f - kg) * ho;
    *hp = hn;
    const half_t hh = (half_t)hn;
    dst0[(size_t)b * ld0 + off0 + s] = hh;
    dst1[(size_t)b * ld1 + off1 + s] = hh;
    if (dst2) dst2[(size_t)b * ld2 + off2 + s] = hh;
  }
}

// Device-wide two-phase barrier (bar[0]=count, bar[1]=generation).
__device__ __forceinline__ void gbar(unsigned* cnt, volatile unsigned* gen) {
  __syncthreads();
  if (threadIdx.x == 0) {
    const unsigned g = *gen;
    __threadfence();
    if (atomicAdd(cnt, 1u) == (unsigned)(NWG - 1)) {
      atomicExch(cnt, 0u);
      __threadfence();
      atomicAdd((unsigned*)gen, 1u);
    } else {
      while (*gen == g) { __builtin_amdgcn_s_sleep(8); }
    }
    __threadfence();
  }
  __syncthreads();
}

// ---------------------------------------------------------------- setup kernels

// Pack all weights into transposed (N x K) zero-padded f16 matrices.
__global__ void pack_kernel(const float* Wjx1, const float* Wjh1,
                            const float* Wkx1, const float* Wkh1,
                            const float* Wjx2, const float* Wjh2,
                            const float* Wkx2, const float* Wkh2,
                            const float* Wwin, const float* Wlin,
                            half_t* Wc1T, half_t* Wc2T,
                            half_t* WwinT, half_t* WlinT) {
  const int N1 = 1024 * D1P;
  const int N2 = 1024 * D2P;
  const int N3 = 32 * CS;
  const int N4 = 128 * CS;
  const int NT = N1 + N2 + N3 + N4;
  for (int i = blockIdx.x * blockDim.x + threadIdx.x; i < NT;
       i += gridDim.x * blockDim.x) {
    float v = 0.0f;
    half_t* dst;
    if (i < N1) {                       // cell1 fused [j|k]:  (83+512) x 1024
      const int n = i / D1P, kk = i % D1P;
      const int s = n & (CS - 1);
      const bool kg = n >= CS;
      if (kk < 83)       v = kg ? Wkx1[kk * CS + s] : Wjx1[kk * CS + s];
      else if (kk < 595) v = kg ? Wkh1[(kk - 83) * CS + s] : Wjh1[(kk - 83) * CS + s];
      dst = Wc1T + i;
    } else if (i < N1 + N2) {           // cell2/3 fused [j|k]: (595+512) x 1024
      const int j = i - N1;
      const int n = j / D2P, kk = j % D2P;
      const int s = n & (CS - 1);
      const bool kg = n >= CS;
      if (kk < 595)       v = kg ? Wkx2[kk * CS + s] : Wjx2[kk * CS + s];
      else if (kk < 1107) v = kg ? Wkh2[(kk - 595) * CS + s] : Wjh2[(kk - 595) * CS + s];
      dst = Wc2T + j;
    } else if (i < N1 + N2 + N3) {      // window head: 512 x 30 -> 32 x 512
      const int j = i - N1 - N2;
      const int n = j / CS, kk = j % CS;
      if (n < 30) v = Wwin[kk * 30 + n];
      dst = WwinT + j;
    } else {                            // output head: 512 x 121 -> 128 x 512
      const int j = i - N1 - N2 - N3;
      const int n = j / CS, kk = j % CS;
      if (n < COUT) v = Wlin[kk * COUT + n];
      dst = WlinT + j;
    }
    *dst = (half_t)v;
  }
}

__global__ void zero_kernel(unsigned long long* p, size_t n) {
  for (size_t i = blockIdx.x * (size_t)blockDim.x + threadIdx.x; i < n;
       i += (size_t)gridDim.x * blockDim.x)
    p[i] = 0ull;
}

// Seed x_0 into the parity-0 activation buffers.
__global__ void x0_kernel(const float* x, half_t* act1, half_t* act2,
                          half_t* act3) {
  const int i = blockIdx.x * blockDim.x + threadIdx.x;
  if (i < CB * 3) {
    const int b = i / 3, d = i % 3;
    const half_t v = (half_t)x[((size_t)b * CT + 0) * 3 + d];
    act1[(size_t)b * D1P + d] = v;
    act2[(size_t)b * D2P + d] = v;
    act3[(size_t)b * D2P + d] = v;
  }
}

// ---------------------------------------------------------------- main scan

struct ScanArgs {
  const float *x, *c_vec;
  const float *bjx1, *bjh1, *bkx1, *bkh1;
  const float *bjx2, *bjh2, *bkx2, *bkh2;
  const float *bwin, *blin;
  const half_t *Wc1T, *Wc2T, *WwinT, *WlinT;
  half_t *act1, *act2, *act3;     // each double-buffered by step parity
  float *h1f32, *h2f32, *h3f32;
  half_t *h1f16, *h3f16;
  float *kappa;
  float *out;
  unsigned *bar;
};

__global__ __launch_bounds__(256) void scan_kernel(ScanArgs A) {
  const int wid  = blockIdx.x * 8 + (threadIdx.x >> 5);  // 0..255 global wave id
  const int lane = threadIdx.x & 31;
  const int tid  = threadIdx.x;

  __shared__ float kgS[16 * 32];
  __shared__ float alS[16 * 10], beS[16 * 10], kaS[16 * 10];
  __shared__ float phiS[16 * 96];

  for (int t = 0; t < CT; ++t) {
    const int p = t & 1;
    half_t* a1c = A.act1 + (size_t)p       * CB * D1P;
    half_t* a1n = A.act1 + (size_t)(1 - p) * CB * D1P;
    half_t* a2c = A.act2 + (size_t)p       * CB * D2P;
    half_t* a2n = A.act2 + (size_t)(1 - p) * CB * D2P;
    half_t* a3c = A.act3 + (size_t)p       * CB * D2P;
    half_t* a3n = A.act3 + (size_t)(1 - p) * CB * D2P;

    // ---- cell 1: [128 x 608] @ [608 x 1024], fused gate epilogue -------------
    {
      const int m0 = (wid >> 5) << 4;
      const int n0 = (wid & 31) << 4;
      v8f cJ = {}, cK = {};
      gemm_pair(a1c, D1P, A.Wc1T, D1P, m0, n0, CS + n0, D1P / 32, lane, cJ, cK);
      gate_epilogue(cJ, cK, A.bjx1, A.bjh1, A.bkx1, A.bkh1, A.h1f32, m0, n0, lane,
                    a1n, D1P, 83,      // h1 -> next step cell1 input
                    a2c, D2P, 3,       // h1 -> this step cell2 input
                    A.h1f16, CS, 0);   // h1 -> window-head A matrix
    }
    gbar(A.bar, (volatile unsigned*)(A.bar + 1));

    // ---- window head + attention (blocks 0..7, 16 batch rows each) -----------
    if (blockIdx.x < 8) {
      const int b0 = blockIdx.x * 16;
      if ((tid >> 5) == 0) {           // wave 0: [16 x 512] @ [512 x 32]
        v8f c0 = {}, c1 = {};
        gemm_pair(A.h1f16, CS, A.WwinT, CS, b0, 0, 16, CS / 32, lane, c0, c1);
        const int n = lane & 15, rb = (lane >> 4) * 8;
        const float bw0 = A.bwin[n];                              // n < 16 < 30
        const float bw1 = (16 + n < 30) ? A.bwin[16 + n] : 0.0f;
#pragma unroll
        for (int v = 0; v < 8; ++v) {
          kgS[(rb + v) * 32 + n]      = c0[v] + bw0;
          kgS[(rb + v) * 32 + 16 + n] = c1[v] + bw1;
        }
      }
      __syncthreads();
      if (tid < 160) {                 // alpha / beta / kappa update
        const int b = tid / 10, k = tid % 10;
        const float al = __expf(kgS[b * 32 + k]);
        const float be = __expf(kgS[b * 32 + 10 + k]);
        float* kp = A.kappa + (size_t)(b0 + b) * CK + k;
        const float ka = *kp + __expf(kgS[b * 32 + 20 + k]);
        *kp = ka;
        alS[b * 10 + k] = al; beS[b * 10 + k] = be; kaS[b * 10 + k] = ka;
      }
      __syncthreads();
      for (int i = tid; i < 16 * 96; i += 256) {   // phi[b,u]
        const int b = i / 96;
        const float u = (float)(i % 96);
        float s = 0.0f;
#pragma unroll
        for (int k = 0; k < 10; ++k) {
          const float d = kaS[b * 10 + k] - u;
          s += alS[b * 10 + k] * __expf(-beS[b * 10 + k] * d * d);
        }
        phiS[i] = s;
      }
      __syncthreads();
      for (int i = tid; i < 16 * 80; i += 256) {   // w[b,c] = sum_u phi * c_vec
        const int b = i / 80, c = i % 80, gb = b0 + b;
        const float* cv = A.c_vec + (size_t)gb * CU * CCD + c;
        float s = 0.0f;
        for (int u = 0; u < CU; ++u) s += phiS[b * 96 + u] * cv[u * CCD];
        const half_t wh = (half_t)s;
        a1n[(size_t)gb * D1P + 3 + c]   = wh;   // next step cell1
        a2c[(size_t)gb * D2P + 515 + c] = wh;   // this step cell2
        a3c[(size_t)gb * D2P + 515 + c] = wh;   // this step cell3
      }
      if (tid < 48 && t + 1 < CT) {    // stage x_{t+1} into next-parity buffers
        const int b = tid / 3, d = tid % 3, gb = b0 + b;
        const half_t xv = (half_t)A.x[((size_t)gb * CT + (t + 1)) * 3 + d];
        a1n[(size_t)gb * D1P + d] = xv;
        a2n[(size_t)gb * D2P + d] = xv;
        a3n[(size_t)gb * D2P + d] = xv;
      }
    }
    gbar(A.bar, (volatile unsigned*)(A.bar + 1));

    // ---- cell 2: [128 x 1120] @ [1120 x 1024] --------------------------------
    {
      const int m0 = (wid >> 5) << 4;
      const int n0 = (wid & 31) << 4;
      v8f cJ = {}, cK = {};
      gemm_pair(a2c, D2P, A.Wc2T, D2P, m0, n0, CS + n0, D2P / 32, lane, cJ, cK);
      gate_epilogue(cJ, cK, A.bjx2, A.bjh2, A.bkx2, A.bkh2, A.h2f32, m0, n0, lane,
                    a2n, D2P, 595,     // h2 recurrent
                    a3c, D2P, 3,       // h2 -> this step cell3 input
                    (half_t*)nullptr, 0, 0);
    }
    gbar(A.bar, (volatile unsigned*)(A.bar + 1));

    // ---- cell 3 (reuses cell2 weights, faithful to source) -------------------
    {
      const int m0 = (wid >> 5) << 4;
      const int n0 = (wid & 31) << 4;
      v8f cJ = {}, cK = {};
      gemm_pair(a3c, D2P, A.Wc2T, D2P, m0, n0, CS + n0, D2P / 32, lane, cJ, cK);
      gate_epilogue(cJ, cK, A.bjx2, A.bjh2, A.bkx2, A.bkh2, A.h3f32, m0, n0, lane,
                    a3n, D2P, 595,     // h3 recurrent
                    A.h3f16, CS, 0,    // h3 -> output-head A matrix
                    (half_t*)nullptr, 0, 0);
    }
    gbar(A.bar, (volatile unsigned*)(A.bar + 1));

    // ---- output head for this step: [128 x 512] @ [512 x 128(pad)] -----------
    if (wid < 64) {
      const int m0 = (wid >> 3) << 4;
      const int n0 = (wid & 7) << 4;
      v8f c = {};
      gemm_one(A.h3f16, CS, A.WlinT, CS, m0, n0, CS / 32, lane, c);
      const int n = lane & 15, rb = (lane >> 4) * 8;
      const int col = n0 + n;
      if (col < COUT) {
        const float bl = A.blin[col];
#pragma unroll
        for (int v = 0; v < 8; ++v) {
          const int b = m0 + rb + v;
          A.out[((size_t)b * CT + t) * COUT + col] = c[v] + bl;
        }
      }
    }
  }
}

// ---------------------------------------------------------------- host launcher

extern "C" void kernel_launch(void* const* d_in, const int* in_sizes, int n_in,
                              void* d_out, int out_size, void* d_ws,
                              size_t ws_size, hipStream_t stream) {
  (void)in_sizes; (void)n_in; (void)out_size; (void)ws_size;

  const float* x     = (const float*)d_in[0];
  const float* c_vec = (const float*)d_in[1];
  const float* Wjx1 = (const float*)d_in[2];  const float* bjx1 = (const float*)d_in[3];
  const float* Wjh1 = (const float*)d_in[4];  const float* bjh1 = (const float*)d_in[5];
  const float* Wkx1 = (const float*)d_in[6];  const float* bkx1 = (const float*)d_in[7];
  const float* Wkh1 = (const float*)d_in[8];  const float* bkh1 = (const float*)d_in[9];
  const float* Wjx2 = (const float*)d_in[10]; const float* bjx2 = (const float*)d_in[11];
  const float* Wjh2 = (const float*)d_in[12]; const float* bjh2 = (const float*)d_in[13];
  const float* Wkx2 = (const float*)d_in[14]; const float* bkx2 = (const float*)d_in[15];
  const float* Wkh2 = (const float*)d_in[16]; const float* bkh2 = (const float*)d_in[17];
  const float* Wwin = (const float*)d_in[18]; const float* bwin = (const float*)d_in[19];
  const float* Wlin = (const float*)d_in[20]; const float* blin = (const float*)d_in[21];

  char* base = (char*)d_ws;
  size_t off = 0;
  auto carve = [&](size_t bytes) -> char* {
    off = (off + 255) & ~(size_t)255;
    char* p = base + off;
    off += bytes;
    return p;
  };

  half_t* Wc1T  = (half_t*)carve((size_t)1024 * D1P * 2);
  half_t* Wc2T  = (half_t*)carve((size_t)1024 * D2P * 2);
  half_t* WwinT = (half_t*)carve((size_t)32 * CS * 2);
  half_t* WlinT = (half_t*)carve((size_t)128 * CS * 2);

  off = (off + 255) & ~(size_t)255;
  const size_t zstart = off;
  half_t* act1  = (half_t*)carve((size_t)2 * CB * D1P * 2);
  half_t* act2  = (half_t*)carve((size_t)2 * CB * D2P * 2);
  half_t* act3  = (half_t*)carve((size_t)2 * CB * D2P * 2);
  float*  h1f32 = (float*)carve((size_t)CB * CS * 4);
  float*  h2f32 = (float*)carve((size_t)CB * CS * 4);
  float*  h3f32 = (float*)carve((size_t)CB * CS * 4);
  half_t* h1f16 = (half_t*)carve((size_t)CB * CS * 2);
  half_t* h3f16 = (half_t*)carve((size_t)CB * CS * 2);
  float*  kappa = (float*)carve((size_t)CB * CK * 4);
  unsigned* bar = (unsigned*)carve(256);
  const size_t zwords = ((off - zstart) + 7) / 8;

  pack_kernel<<<dim3(1024), dim3(256), 0, stream>>>(
      Wjx1, Wjh1, Wkx1, Wkh1, Wjx2, Wjh2, Wkx2, Wkh2, Wwin, Wlin,
      Wc1T, Wc2T, WwinT, WlinT);
  zero_kernel<<<dim3(512), dim3(256), 0, stream>>>(
      (unsigned long long*)(base + zstart), zwords);
  x0_kernel<<<dim3(2), dim3(256), 0, stream>>>(x, act1, act2, act3);

  ScanArgs args;
  args.x = x; args.c_vec = c_vec;
  args.bjx1 = bjx1; args.bjh1 = bjh1; args.bkx1 = bkx1; args.bkh1 = bkh1;
  args.bjx2 = bjx2; args.bjh2 = bjh2; args.bkx2 = bkx2; args.bkh2 = bkh2;
  args.bwin = bwin; args.blin = blin;
  args.Wc1T = Wc1T; args.Wc2T = Wc2T; args.WwinT = WwinT; args.WlinT = WlinT;
  args.act1 = act1; args.act2 = act2; args.act3 = act3;
  args.h1f32 = h1f32; args.h2f32 = h2f32; args.h3f32 = h3f32;
  args.h1f16 = h1f16; args.h3f16 = h3f16;
  args.kappa = kappa;
  args.out = (float*)d_out;
  args.bar = bar;

  scan_kernel<<<dim3(NWG), dim3(256), 0, stream>>>(args);
}
